// RSSM_89429809037910
// MI455X (gfx1250) — matrix operand order repaired and checked by reading, not compiled, not run on hardware
//
#include <hip/hip_runtime.h>

// ---------------------------------------------------------------------------
// RSSM scan for MI455X (gfx1250): bf16 WMMA 16x16x32, persistent per-batch-tile
// workgroups, LDS-resident recurrent state, prepacked transposed bf16 weights.
// All GEMMs: compute 16x16 tile -> immediate write to the *other* LDS buffer,
// so no accumulators live across barriers (no scratch spills).
// ---------------------------------------------------------------------------

#define DEVINL __device__ __forceinline__

typedef __attribute__((ext_vector_type(16))) __bf16        bf16x16;
typedef __attribute__((ext_vector_type(8)))  float         f32x8;
typedef __attribute__((ext_vector_type(4)))  unsigned int  u32x4;

// Problem sizes
constexpr int Bsz = 192, Tsz = 64;
constexpr int ACT = 16, EMB = 1024, STOCH = 32, DETER = 600, HID = 600;

// Padded GEMM shapes (K padded to mult of 32, N padded to mult of 16)
constexpr int KP1  = 64,   NP1  = 608;   // img1: K=48,   N=600
constexpr int KPG  = 1216, NPG  = 1808;  // gru : K=1200, N=1800
constexpr int KP2  = 608,  NP2  = 608;   // img2: K=600,  N=600
constexpr int KP3  = 608,  NP3  = 64;    // img3: K=600,  N=64
constexpr int KPO1 = 1632, NPO1 = 608;   // obs1: K=1624, N=600
constexpr int KPO2 = 608,  NPO2 = 64;    // obs2: K=600,  N=64

// Workspace layout (ushort units): transposed padded bf16 weights [Npad][Kpad]
constexpr size_t OFF1  = 0;
constexpr size_t OFFG  = OFF1  + (size_t)NP1  * KP1;
constexpr size_t OFF2  = OFFG  + (size_t)NPG  * KPG;
constexpr size_t OFF3  = OFF2  + (size_t)NP2  * KP2;
constexpr size_t OFFO1 = OFF3  + (size_t)NP3  * KP3;
constexpr size_t OFFO2 = OFFO1 + (size_t)NPO1 * KPO1;

// LDS layout (byte offsets), all chunks 16B aligned
constexpr int LDA = 1632;                 // A-staging row stride (ushort)
constexpr int LDO = 1808;                 // output-staging row stride (ushort)
constexpr int SM_BUFA = 0;                              // ushort[16*1632]
constexpr int SM_BUFO = SM_BUFA + 16 * LDA * 2;         // ushort[16*1808]
constexpr int SM_DET  = SM_BUFO + 16 * LDO * 2;         // ushort[16*608] (bf16 deter state)
constexpr int SM_STO  = SM_DET  + 16 * 608 * 2;         // ushort[16*32]  (bf16 stoch state)
constexpr int SM_RED  = SM_STO  + 16 * 32 * 2;          // float[512]     (LN partials)
constexpr int SM_STAT = SM_RED  + 512 * 4;              // float[32]      (mean, rstd)
constexpr int SM_TMP  = SM_STAT + 32 * 4;               // float[16*64]   (mean/std staging)
constexpr int SM_TOTAL = SM_TMP + 16 * 64 * 4;          // 136832 B

// ---------------- scalar helpers ----------------
DEVINL unsigned short f2bf(float f) {
  union { float f; unsigned u; } v; v.f = f;
  unsigned r = (v.u + 0x7FFFu + ((v.u >> 16) & 1u)) >> 16;   // RNE
  return (unsigned short)r;
}
DEVINL float bf2f(unsigned short h) {
  union { unsigned u; float f; } v; v.u = ((unsigned)h) << 16; return v.f;
}
DEVINL float eluf(float x)  { return x > 0.f ? x : (__expf(x) - 1.f); }
DEVINL float sigm(float x)  { return 1.f / (1.f + __expf(-x)); }
DEVINL float stdfn(float r) { return 2.f * sigm(0.5f * r) + 0.1f; }   // 2*sigmoid(raw/2)+min_std

// ---------------- WMMA tile loaders (ISA 7.12.2 layouts) ----------------
// A (16x32 bf16): lane l -> row m=l&15, hi=l>>4; halfs = [hi*8+0..7] and [16+hi*8+0..7]
DEVINL bf16x16 ld_tileA(const unsigned short* p) {
  union { u32x4 q[2]; bf16x16 v; } u;
  u.q[0] = *(const u32x4*)(p);
  u.q[1] = *(const u32x4*)(p + 16);
  return u.v;
}
// B (32x16 bf16) from transposed weights [Npad][Kpad]: lane l -> col n=l&15,
// K = (l>>4)*8 + 0..7 and 16 + (l>>4)*8 + 0..7 (same order as A)
DEVINL bf16x16 ld_tileB(const unsigned short* p) {
  union { u32x4 q[2]; bf16x16 v; } u;
  u.q[0] = *(const u32x4*)(p);
  u.q[1] = *(const u32x4*)(p + 16);
  return u.v;
}

// One 16x16 output tile: sA is LDS activations (row-major, stride lda),
// wT is global transposed bf16 weight [Npad][Kpad].
DEVINL f32x8 wmma_tile(const unsigned short* sA, int lda,
                       const unsigned short* wT, int kp,
                       int ntile, int ktiles, int lane) {
  const int m  = lane & 15;
  const int hi = lane >> 4;
  const unsigned short* aB = sA + m * lda + hi * 8;
  const unsigned short* bB = wT + (size_t)(ntile * 16 + m) * kp + hi * 8;
  f32x8 acc = {0.f, 0.f, 0.f, 0.f, 0.f, 0.f, 0.f, 0.f};
  for (int kt = 0; kt < ktiles; ++kt) {
    __builtin_prefetch(bB + kt * 32 + 64, 0, 1);   // global_prefetch_b8 on B stream
    bf16x16 a = ld_tileA(aB + kt * 32);
    bf16x16 b = ld_tileB(bB + kt * 32);
    acc = __builtin_amdgcn_wmma_f32_16x16x32_bf16(false, a, false, b,
                                                  (short)0, acc, false, false);
  }
  return acc;
}

// ---------------- weight prepack: fp32 [K][N] -> bf16 transposed [Npad][Kpad] ----
__global__ void prep_weight(const float* __restrict__ w, unsigned short* __restrict__ wt,
                            int K, int N, int Kp, int Np) {
  int i = blockIdx.x * blockDim.x + threadIdx.x;
  int tot = Kp * Np;
  if (i >= tot) return;
  int n = i / Kp, k = i % Kp;
  float v = (n < N && k < K) ? w[(size_t)k * N + n] : 0.f;
  wt[i] = f2bf(v);
}

// ---------------- main persistent scan kernel ----------------
// grid = 12 blocks (16 batch rows each), block = 256 threads = 8 wave32s.
__global__ void rssm_scan(const float* __restrict__ action,
                          const float* __restrict__ embed,
                          const float* __restrict__ noise_pr,
                          const float* __restrict__ noise_po,
                          const float* __restrict__ b1,
                          const float* __restrict__ bg,
                          const float* __restrict__ lnsc,
                          const float* __restrict__ lnbi,
                          const float* __restrict__ b2,
                          const float* __restrict__ b3,
                          const float* __restrict__ bo1,
                          const float* __restrict__ bo2,
                          const unsigned short* __restrict__ ws,
                          float* __restrict__ out) {
  extern __shared__ unsigned char smem[];
  unsigned short* bufA  = (unsigned short*)(smem + SM_BUFA);
  unsigned short* bufO  = (unsigned short*)(smem + SM_BUFO);
  unsigned short* deterS= (unsigned short*)(smem + SM_DET);
  unsigned short* stochS= (unsigned short*)(smem + SM_STO);
  float* sRed  = (float*)(smem + SM_RED);
  float* sStat = (float*)(smem + SM_STAT);
  float* sTmp  = (float*)(smem + SM_TMP);

  const int tid  = threadIdx.x;
  const int wave = tid >> 5;
  const int lane = tid & 31;
  const int m    = lane & 15;
  const int hi   = lane >> 4;
  const int gr0  = blockIdx.x * 16;       // batch row base

  const unsigned short* WT1  = ws + OFF1;
  const unsigned short* WTG  = ws + OFFG;
  const unsigned short* WT2  = ws + OFF2;
  const unsigned short* WT3  = ws + OFF3;
  const unsigned short* WTO1 = ws + OFFO1;
  const unsigned short* WTO2 = ws + OFFO2;

  // init recurrent state = 0
  for (int i = tid; i < 16 * 608; i += 256) deterS[i] = 0;
  for (int i = tid; i < 16 * 32;  i += 256) stochS[i] = 0;
  __syncthreads();

  for (int t = 0; t < Tsz; ++t) {
    // ---- stage A1 = [stoch | action | pad] into bufA[16][64] ----
    for (int i = tid; i < 16 * 64; i += 256) {
      int r = i >> 6, c = i & 63;
      unsigned short v;
      if (c < STOCH)            v = stochS[r * 32 + c];
      else if (c < STOCH + ACT) v = f2bf(action[((size_t)(gr0 + r) * Tsz + t) * ACT + (c - STOCH)]);
      else                      v = 0;
      bufA[r * LDA + c] = v;
    }
    __syncthreads();

    // ---- G1: x = elu(A1 @ W1 + b1), N=600 -> bufO (bufO is dead here) ----
    for (int nt = wave; nt < 38; nt += 8) {
      f32x8 acc = wmma_tile(bufA, LDA, WT1, KP1, nt, KP1 / 32, lane);
      #pragma unroll
      for (int v = 0; v < 8; ++v) {
        int col = nt * 16 + m;
        if (col < HID) {
          int row = hi * 8 + v;
          bufO[row * LDO + col] = f2bf(eluf(acc[v] + b1[col]));
        }
      }
    }
    __syncthreads();

    // ---- stage A2 = [x | deter | 0] into bufA[16][1216] ----
    for (int i = tid; i < 16 * 1216; i += 256) {
      int r = i / 1216, j = i % 1216;
      unsigned short v;
      if (j < HID)            v = bufO[r * LDO + j];
      else if (j < HID + DETER) v = deterS[r * 608 + (j - HID)];
      else                    v = 0;
      bufA[r * LDA + j] = v;
    }
    __syncthreads();

    // ---- G2: parts_raw = A2 @ Wg + bg, N=1800 -> bufO (bf16) ----
    for (int nt = wave; nt < 113; nt += 8) {
      f32x8 acc = wmma_tile(bufA, LDA, WTG, KPG, nt, KPG / 32, lane);
      #pragma unroll
      for (int v = 0; v < 8; ++v) {
        int col = nt * 16 + m;
        if (col < 1800) {
          int row = hi * 8 + v;
          bufO[row * LDO + col] = f2bf(acc[v] + bg[col]);
        }
      }
    }
    __syncthreads();

    // ---- LayerNorm stats over 1800 per row ----
    {
      int r = tid >> 4, sub = tid & 15;
      float s = 0.f, ss = 0.f;
      for (int c = sub; c < 1800; c += 16) {
        float x = bf2f(bufO[r * LDO + c]);
        s += x; ss += x * x;
      }
      sRed[tid] = s; sRed[256 + tid] = ss;
      __syncthreads();
      if (tid < 16) {
        float S = 0.f, SS = 0.f;
        for (int j = 0; j < 16; ++j) { S += sRed[tid * 16 + j]; SS += sRed[256 + tid * 16 + j]; }
        float mean = S / 1800.f;
        float var  = SS / 1800.f - mean * mean;
        sStat[tid * 2]     = mean;
        sStat[tid * 2 + 1] = rsqrtf(var + 1e-5f);
      }
      __syncthreads();
    }

    // ---- GRU blend -> deter_n (state + A cols 0..599 + outputs), stage embed ----
    for (int i = tid; i < 16 * DETER; i += 256) {
      int r = i / DETER, c = i % DETER;
      float mean = sStat[r * 2], rstd = sStat[r * 2 + 1];
      float xr = (bf2f(bufO[r * LDO + c])        - mean) * rstd * lnsc[c]        + lnbi[c];
      float xc = (bf2f(bufO[r * LDO + 600 + c])  - mean) * rstd * lnsc[600 + c]  + lnbi[600 + c];
      float xu = (bf2f(bufO[r * LDO + 1200 + c]) - mean) * rstd * lnsc[1200 + c] + lnbi[1200 + c];
      float reset = sigm(xr);
      float cand  = tanhf(reset * xc);
      float upd   = sigm(xu - 1.f);
      float dn = upd * cand + (1.f - upd) * bf2f(deterS[r * 608 + c]);
      deterS[r * 608 + c] = f2bf(dn);
      bufA[r * LDA + c]   = f2bf(dn);
      float* ob = out + ((size_t)(gr0 + r) * Tsz + t) * 1392;
      ob[96 + c]  = dn;
      ob[792 + c] = dn;
    }
    for (int i = tid; i < 16 * 1032; i += 256) {     // cols 600..1631: embed + pad
      int r = i / 1032, j = i % 1032;
      float v = (j < EMB) ? embed[((size_t)(gr0 + r) * Tsz + t) * EMB + j] : 0.f;
      bufA[r * LDA + 600 + j] = f2bf(v);
    }
    __syncthreads();

    // ---- G5: obs1 = elu([deter_n|embed] @ Wo1 + bo1) -> bufO ----
    for (int nt = wave; nt < 38; nt += 8) {
      f32x8 acc = wmma_tile(bufA, LDA, WTO1, KPO1, nt, KPO1 / 32, lane);
      #pragma unroll
      for (int v = 0; v < 8; ++v) {
        int col = nt * 16 + m;
        int row = hi * 8 + v;
        float val = (col < HID) ? eluf(acc[v] + bo1[col]) : 0.f;
        bufO[row * LDO + col] = f2bf(val);
      }
    }
    __syncthreads();

    // ---- G6: obs2 -> mean_po | std_po into sTmp ----
    if (wave < 4) {
      f32x8 acc = wmma_tile(bufO, LDO, WTO2, KPO2, wave, KPO2 / 32, lane);
      #pragma unroll
      for (int v = 0; v < 8; ++v) {
        int col = wave * 16 + m;
        int row = hi * 8 + v;
        float val = acc[v] + bo2[col];
        if (col >= STOCH) val = stdfn(val);
        sTmp[row * 64 + col] = val;
      }
    }
    __syncthreads();
    // finalize posterior: stoch_po = mean + std*noise; update stoch state
    for (int i = tid; i < 16 * STOCH; i += 256) {
      int r = i / STOCH, c = i % STOCH;
      float mean = sTmp[r * 64 + c];
      float sd   = sTmp[r * 64 + 32 + c];
      float st   = mean + sd * noise_po[((size_t)(gr0 + r) * Tsz + t) * STOCH + c];
      stochS[r * 32 + c] = f2bf(st);
      float* ob = out + ((size_t)(gr0 + r) * Tsz + t) * 1392;
      ob[c] = mean; ob[32 + c] = sd; ob[64 + c] = st;
    }
    // zero A3 pad cols 600..607 (they currently hold embed)
    for (int i = tid; i < 16 * 8; i += 256) {
      int r = i >> 3, j = i & 7;
      bufA[r * LDA + 600 + j] = 0;
    }
    __syncthreads();

    // ---- G3: img2 = elu(deter_n @ W2 + b2) -> bufO ----
    for (int nt = wave; nt < 38; nt += 8) {
      f32x8 acc = wmma_tile(bufA, LDA, WT2, KP2, nt, KP2 / 32, lane);
      #pragma unroll
      for (int v = 0; v < 8; ++v) {
        int col = nt * 16 + m;
        int row = hi * 8 + v;
        float val = (col < HID) ? eluf(acc[v] + b2[col]) : 0.f;
        bufO[row * LDO + col] = f2bf(val);
      }
    }
    __syncthreads();

    // ---- G4: img3 -> mean_pr | std_pr into sTmp ----
    if (wave < 4) {
      f32x8 acc = wmma_tile(bufO, LDO, WT3, KP3, wave, KP3 / 32, lane);
      #pragma unroll
      for (int v = 0; v < 8; ++v) {
        int col = wave * 16 + m;
        int row = hi * 8 + v;
        float val = acc[v] + b3[col];
        if (col >= STOCH) val = stdfn(val);
        sTmp[row * 64 + col] = val;
      }
    }
    __syncthreads();
    // finalize prior: stoch_pr = mean + std*noise (output only, not state)
    for (int i = tid; i < 16 * STOCH; i += 256) {
      int r = i / STOCH, c = i % STOCH;
      float mean = sTmp[r * 64 + c];
      float sd   = sTmp[r * 64 + 32 + c];
      float st   = mean + sd * noise_pr[((size_t)(gr0 + r) * Tsz + t) * STOCH + c];
      float* ob = out + ((size_t)(gr0 + r) * Tsz + t) * 1392;
      ob[696 + c] = mean; ob[728 + c] = sd; ob[760 + c] = st;
    }
    __syncthreads();
  }
}

// ---------------------------------------------------------------------------
extern "C" void kernel_launch(void* const* d_in, const int* in_sizes, int n_in,
                              void* d_out, int out_size, void* d_ws, size_t ws_size,
                              hipStream_t stream) {
  (void)in_sizes; (void)n_in; (void)out_size; (void)ws_size;
  const float* action   = (const float*)d_in[0];
  const float* embed    = (const float*)d_in[1];
  const float* noise_pr = (const float*)d_in[2];
  const float* noise_po = (const float*)d_in[3];
  const float* img1_w   = (const float*)d_in[4];
  const float* img1_b   = (const float*)d_in[5];
  const float* gru_w    = (const float*)d_in[6];
  const float* gru_b    = (const float*)d_in[7];
  const float* ln_scale = (const float*)d_in[8];
  const float* ln_bias  = (const float*)d_in[9];
  const float* img2_w   = (const float*)d_in[10];
  const float* img2_b   = (const float*)d_in[11];
  const float* img3_w   = (const float*)d_in[12];
  const float* img3_b   = (const float*)d_in[13];
  const float* obs1_w   = (const float*)d_in[14];
  const float* obs1_b   = (const float*)d_in[15];
  const float* obs2_w   = (const float*)d_in[16];
  const float* obs2_b   = (const float*)d_in[17];
  unsigned short* ws = (unsigned short*)d_ws;
  float* outp = (float*)d_out;

  auto blocks = [](size_t n) { return (unsigned)((n + 255) / 256); };
  // prepack weights (runs every call: deterministic, ws not assumed persistent)
  prep_weight<<<blocks((size_t)NP1  * KP1 ), 256, 0, stream>>>(img1_w, ws + OFF1,  48,   HID,  KP1,  NP1);
  prep_weight<<<blocks((size_t)NPG  * KPG ), 256, 0, stream>>>(gru_w,  ws + OFFG,  1200, 1800, KPG,  NPG);
  prep_weight<<<blocks((size_t)NP2  * KP2 ), 256, 0, stream>>>(img2_w, ws + OFF2,  HID,  HID,  KP2,  NP2);
  prep_weight<<<blocks((size_t)NP3  * KP3 ), 256, 0, stream>>>(img3_w, ws + OFF3,  HID,  64,   KP3,  NP3);
  prep_weight<<<blocks((size_t)NPO1 * KPO1), 256, 0, stream>>>(obs1_w, ws + OFFO1, 1624, HID,  KPO1, NPO1);
  prep_weight<<<blocks((size_t)NPO2 * KPO2), 256, 0, stream>>>(obs2_w, ws + OFFO2, HID,  64,   KPO2, NPO2);

  (void)hipFuncSetAttribute(reinterpret_cast<const void*>(rssm_scan),
                            hipFuncAttributeMaxDynamicSharedMemorySize, SM_TOTAL);
  rssm_scan<<<Bsz / 16, 256, SM_TOTAL, stream>>>(
      action, embed, noise_pr, noise_po,
      img1_b, gru_b, ln_scale, ln_bias, img2_b, img3_b, obs1_b, obs2_b,
      ws, outp);
}
